// NePuRenderer_28286654612092
// MI455X (gfx1250) — compile-verified, matrix operand order checked by reading, not run on hardware
//
#include <hip/hip_runtime.h>
#include <stdint.h>
#include <stddef.h>

// ---------------------------------------------------------------------------
// NePu renderer for MI455X (gfx1250, wave32, WMMA).
//
//   K_prep0    : lat = z@down_W+b ; kps = [kps_2d, depth-min] (1 block)
//   K_pack     : repack ALL cross/head weights to f16 WMMA B-fragment layout
//                (pe_W gets bias folded into padded K-row 33)
//   K_tab/K_tattn/K_bn (x3) : transformer blocks over 64 anchors (scalar f32)
//   K_crossprep: K/V tables (f16), q/kg/vg vectors, cross biases
//   K_attn     : cross-attention; wave = 4 queries x 8 slots = 2 WMMA M-tiles,
//                each B fragment reused for both tiles (halves L2 traffic)
//   K_heads    : 3 MLP heads, 32 queries (2 M-tiles) per wave; next-layer
//                weight prefetch via global_prefetch_b8
//
// Input order assumption (jax pytree recursive-alphabetical after 6 arrays):
//   color head 6..39, ct1 40..50, depth 51..84, down_W 85, down_b 86,
//   occ 87..120, t1 121..131, t2 132..142, t3 143..153.
// ---------------------------------------------------------------------------

typedef _Float16 h16;
typedef __attribute__((ext_vector_type(16))) _Float16 v16h;
typedef __attribute__((ext_vector_type(8)))  _Float16 v8h;
typedef __attribute__((ext_vector_type(8)))  float    v8f;

constexpr int B_ = 2, Q_ = 32768, NA_ = 64, LAT_ = 256;
constexpr int NN_ = 7;

// ---------------- workspace layout (bytes) ----------------
constexpr size_t OFF_LAT = 0;                          // B*128 f32
constexpr size_t OFF_KPS = 1024;                       // B*64*3 f32
constexpr size_t OFF_FA  = 4096;                       // feats A: B*64*128 f32
constexpr size_t OFF_FB  = OFF_FA + 65536;             // feats B (raw res)
constexpr size_t OFF_BN  = OFF_FB + 65536;             // sum[128],sumsq[128]
constexpr size_t OFF_QT  = OFF_BN + 1024;              // qtab  B*64*128 f32
constexpr size_t OFF_KT  = OFF_QT + 65536;
constexpr size_t OFF_VT  = OFF_KT + 65536;
constexpr size_t OFF_KTH = OFF_VT + 65536;             // K table f16
constexpr size_t OFF_VTH = OFF_KTH + 32768;            // V table f16
constexpr size_t OFF_QV  = OFF_VTH + 32768;            // q vec B*128 f32
constexpr size_t OFF_KG  = OFF_QV + 1024;
constexpr size_t OFF_VG  = OFF_KG + 1024;
constexpr size_t OFF_CB  = OFF_VG + 1024;              // g_b1,g_b2 f32
constexpr size_t OFF_PEW = ((OFF_CB + 1536 + 255) / 256) * 256;  // 16 frags
constexpr size_t OFF_W1P = OFF_PEW + 16 * 1024;        // 32 frags
constexpr size_t OFF_W2P = OFF_W1P + 32 * 1024;        // 32 frags
constexpr size_t OFF_HEAD = OFF_W2P + 32 * 1024;       // 3 heads
constexpr size_t HEAD_STRIDE = 192 * 1024;
constexpr size_t HB_BIAS = 182272;                     // byte off of f32 biases
constexpr size_t OFF_LOC = ((OFF_HEAD + 3 * HEAD_STRIDE + 4095) / 4096) * 4096;

struct Ptrs { const float* p[160]; };

// ---------------- WMMA helpers ----------------
__device__ inline v8f wmma(v16h a, v16h b, v8f c) {
  return __builtin_amdgcn_wmma_f32_16x16x32_f16(false, a, false, b,
                                                (short)0, c, false, false);
}

union AF { v16h v; unsigned u[8]; h16 h[16]; };

// A fragment (16 rows x 32 K) from row-major f16 (LDS or global), ISA layout:
// lane L: M=L&15, group g=L>>4; pair p: K = (p&3)*2 + g*8 + (p>=4?16:0)
template <typename T>
__device__ inline v16h load_A(const T* X, int ldk, int k0) {
  int lane = threadIdx.x & 31;
  int m = lane & 15, g = lane >> 4;
  AF a;
#pragma unroll
  for (int p = 0; p < 8; ++p) {
    int kp = ((p & 3) * 2) + (g ? 8 : 0) + ((p >= 4) ? 16 : 0);
    a.u[p] = *(const unsigned*)(X + m * ldk + k0 + kp);
  }
  return a.v;
}

// B fragment from packed store: frag = 32 lanes x 16 halves contiguous (1KB)
__device__ inline v16h load_B_frag(const h16* packed, int f) {
  const h16* p = packed + (size_t)f * 512 + (threadIdx.x & 31) * 16;
  union { v16h v; v8h x[2]; } b;
  b.x[0] = *(const v8h*)p;
  b.x[1] = *(const v8h*)(p + 8);
  return b.v;
}

// per-lane prefetch of a fragment region (global_prefetch_b8 hints into L0/L2)
__device__ inline void prefetch_frags(const h16* p, int bytes) {
  int lane = threadIdx.x & 31;
  for (int o = lane * 128; o < bytes; o += 32 * 128)
    __builtin_prefetch((const char*)p + o, 0, 0);
}

// f32 D tile (16x16) -> row-major f16; bias added per column, optional relu
__device__ inline void store_tile(h16* X, int ldn, int n0, v8f d, float bias,
                                  bool relu) {
  int lane = threadIdx.x & 31;
  int n = n0 + (lane & 15), mb = (lane >> 4) * 8;
#pragma unroll
  for (int r = 0; r < 8; ++r) {
    float v = d[r] + bias;
    if (relu) v = fmaxf(v, 0.f);
    X[(mb + r) * ldn + n] = (h16)v;
  }
}

// no-bias variant (avoids literal +0.0 adds)
__device__ inline void store_tile0(h16* X, int ldn, int n0, v8f d, bool relu) {
  int lane = threadIdx.x & 31;
  int n = n0 + (lane & 15), mb = (lane >> 4) * 8;
#pragma unroll
  for (int r = 0; r < 8; ++r) {
    float v = d[r];
    if (relu) v = fmaxf(v, 0.f);
    X[(mb + r) * ldn + n] = (h16)v;
  }
}

// positional embedding: [pd, sin(pd*f), cos(pd*f) for f in linspace(1,32,5)]
__device__ inline void posembed(float dx, float dy, float dz, float* e) {
  const float FR[5] = {1.f, 8.75f, 16.5f, 24.25f, 32.f};
  e[0] = dx; e[1] = dy; e[2] = dz;
#pragma unroll
  for (int i = 0; i < 5; ++i) {
    float f = FR[i];
    e[3 + 6 * i + 0] = __sinf(dx * f);
    e[3 + 6 * i + 1] = __sinf(dy * f);
    e[3 + 6 * i + 2] = __sinf(dz * f);
    e[6 + 6 * i + 0] = __cosf(dx * f);
    e[6 + 6 * i + 1] = __cosf(dy * f);
    e[6 + 6 * i + 2] = __cosf(dz * f);
  }
}

// pack Wt (from W: N x K row-major) into B fragments, zero-padded
__device__ void pack(h16* dst, const float* W, int N, int K, int NT, int KT,
                     int tid, int nthr) {
  int total = NT * KT * 512;
  for (int i = tid; i < total; i += nthr) {
    int f = i >> 9, r = i & 511;
    int lane = r >> 4, hh = r & 15;
    int nt = f / KT, kt = f % KT;
    int g = lane >> 4, dn = lane & 15;
    int dk = ((hh >= 8) ? 16 : 0) + g * 8 + (hh & 7);
    int k = kt * 32 + dk, n = nt * 16 + dn;
    dst[i] = (h16)((k < K && n < N) ? W[n * K + k] : 0.f);
  }
}

// pe_W pack with bias riding padded K-row 33 (A supplies 1.0 there)
__device__ void packPE(h16* dst, const float* W, const float* bias,
                       int tid, int nthr) {
  const int NT = 8, KT = 2, N = 128, K = 33;
  int total = NT * KT * 512;
  for (int i = tid; i < total; i += nthr) {
    int f = i >> 9, r = i & 511;
    int lane = r >> 4, hh = r & 15;
    int nt = f / KT, kt = f % KT;
    int g = lane >> 4, dn = lane & 15;
    int dk = ((hh >= 8) ? 16 : 0) + g * 8 + (hh & 7);
    int k = kt * 32 + dk, n = nt * 16 + dn;
    float v = 0.f;
    if (n < N) {
      if (k < K) v = W[n * K + k];
      else if (k == K) v = bias[n];
    }
    dst[i] = (h16)v;
  }
}

// ---------------- K_prep0: lat + keypoints ----------------
__global__ __launch_bounds__(128) void K_prep0(
    const float* z, const float* dW, const float* db, const float* anchors,
    const float* extr, const float* kps2d, uint8_t* wsb) {
  __shared__ float dbuf[2][64];
  int tid = threadIdx.x;
  float* lat = (float*)(wsb + OFF_LAT);
  for (int o = tid; o < 256; o += 128) {
    int b = o >> 7, f = o & 127;
    float acc = db[f];
    for (int d = 0; d < LAT_; ++d) acc += z[b * LAT_ + d] * dW[f * LAT_ + d];
    lat[o] = acc;
  }
  {
    int b = tid >> 6, m = tid & 63;
    const float* A = anchors + ((size_t)b * 64 + m) * 3;
    const float* E = extr + b * 12 + 8;  // row 2 of extrinsics
    dbuf[b][m] = E[0] * A[0] + E[1] * A[1] + E[2] * A[2] + E[3];
  }
  __syncthreads();
  {
    int b = tid >> 6, m = tid & 63;
    float mn = 3.4e38f;
    for (int a = 0; a < 64; ++a) mn = fminf(mn, dbuf[b][a]);
    float* kp = (float*)(wsb + OFF_KPS) + ((size_t)b * 64 + m) * 3;
    kp[0] = kps2d[(b * 64 + m) * 2 + 0];
    kp[1] = kps2d[(b * 64 + m) * 2 + 1];
    kp[2] = dbuf[b][m] - mn;
  }
}

// ---------------- K_pack: weight repack to fragment f16 ----------------
__global__ __launch_bounds__(256) void K_pack(Ptrs P, uint8_t* wsb) {
  int gtid = blockIdx.x * 256 + threadIdx.x;
  int gsz = gridDim.x * 256;
  packPE((h16*)(wsb + OFF_PEW), P.p[44], P.p[45], gtid, gsz);       // pe_W+b
  pack((h16*)(wsb + OFF_W1P), P.p[40], 128, 128, 8, 4, gtid, gsz);  // g_W1
  pack((h16*)(wsb + OFF_W2P), P.p[41], 128, 128, 8, 4, gtid, gsz);  // g_W2
  const int hbase[3] = {87, 51, 6};   // occ, depth, color
  const int odims[3] = {1, 1, 3};
  for (int s = 0; s < 3; ++s) {
    int hb = hbase[s];
    h16* base = (h16*)(wsb + OFF_HEAD + (size_t)s * HEAD_STRIDE);
    pack(base, P.p[hb + 30], 64, 128, 4, 4, gtid, gsz);             // init_W
    for (int i = 0; i < 5; ++i)
      pack(base + 8192 + 8192 * i, P.p[hb + 20 + 2 * i], 64, 128, 4, 4,
           gtid, gsz);                                              // fc_c W
    for (int i = 0; i < 5; ++i) {
      pack(base + 49152 + 8192 * i, P.p[hb + 4 * i + 0], 64, 64, 4, 2,
           gtid, gsz);                                              // W0
      pack(base + 49152 + 8192 * i + 4096, P.p[hb + 4 * i + 1], 64, 64, 4, 2,
           gtid, gsz);                                              // W1
    }
    pack(base + 90112, P.p[hb + 32], odims[s], 64, 1, 2, gtid, gsz); // out_W
    float* bias = (float*)(wsb + OFF_HEAD + (size_t)s * HEAD_STRIDE + HB_BIAS);
    for (int j = gtid; j < 1040; j += gsz) {
      float v;
      if (j < 64) v = P.p[hb + 31][j];                              // init_b
      else if (j < 384) { int i = (j - 64) >> 6; v = P.p[hb + 21 + 2 * i][(j - 64) & 63]; }
      else if (j < 1024) {
        int i = (j - 384) >> 7, r = (j - 384) & 127;
        v = (r < 64) ? P.p[hb + 4 * i + 2][r] : P.p[hb + 4 * i + 3][r - 64];
      } else { int c = j - 1024; v = (c < odims[s]) ? P.p[hb + 33][c] : 0.f; }
      bias[j] = v;
    }
  }
}

// ---------------- transformer-block helpers (scalar, tiny) ----------------
__global__ __launch_bounds__(256) void K_tab(const float* featsIn,
    const float* wqs, const float* wks, const float* wvs, uint8_t* wsb) {
  int b = blockIdx.x, tid = threadIdx.x;
  __shared__ float sf[8192];
  const float* F = featsIn + (size_t)b * 8192;
  for (int i = tid; i < 8192; i += 256) sf[i] = F[i];
  float* bn = (float*)(wsb + OFF_BN);
  for (int i = tid; i < 256; i += 256) bn[i] = 0.f;
  __syncthreads();
  float* qt = (float*)(wsb + OFF_QT) + (size_t)b * 8192;
  float* kt = (float*)(wsb + OFF_KT) + (size_t)b * 8192;
  float* vt = (float*)(wsb + OFF_VT) + (size_t)b * 8192;
  for (int i = tid; i < 8192; i += 256) {
    int m = i >> 7, f = i & 127;
    float a1 = 0, a2 = 0, a3 = 0;
    for (int d = 0; d < 128; ++d) {
      float x = sf[m * 128 + d];
      a1 += x * wqs[f * 128 + d];
      a2 += x * wks[f * 128 + d];
      a3 += x * wvs[f * 128 + d];
    }
    qt[i] = a1; kt[i] = a2; vt[i] = a3;
  }
}

__global__ __launch_bounds__(128) void K_tattn(
    const float* featsIn, const float* peW, const float* peb,
    const float* W1, const float* b1, const float* W2, const float* b2,
    uint8_t* wsb, float* resOut, int knnK) {
  __shared__ h16 pe[64 * 128], tb[64 * 128], hbuf[64 * 128];
  __shared__ int idx[64];
  __shared__ float skps[64 * 3];
  int bm = blockIdx.x, b = bm >> 6, m = bm & 63;
  int tid = threadIdx.x;
  const float* kps = (const float*)(wsb + OFF_KPS) + (size_t)b * 192;
  for (int i = tid; i < 192; i += 128) skps[i] = kps[i];
  __syncthreads();
  int n = knnK;
  if (knnK == 64) {
    for (int i = tid; i < 64; i += 128) idx[i] = i;
  } else if (tid == 0) {
    float bd[8]; int bi[8];
    for (int j = 0; j < 8; ++j) { bd[j] = 3.4e38f; bi[j] = 0; }
    float x0 = skps[m * 3], x1 = skps[m * 3 + 1], x2 = skps[m * 3 + 2];
    for (int a = 0; a < 64; ++a) {
      float dx = x0 - skps[a * 3], dy = x1 - skps[a * 3 + 1],
            dz = x2 - skps[a * 3 + 2];
      float d2 = dx * dx + dy * dy + dz * dz;
      if (d2 < bd[7]) {
        int j = 7;
        while (j > 0 && bd[j - 1] > d2) { bd[j] = bd[j - 1]; bi[j] = bi[j - 1]; --j; }
        bd[j] = d2; bi[j] = a;
      }
    }
    for (int j = 0; j < 8; ++j) idx[j] = bi[j];
  }
  __syncthreads();
  const float* qt = (const float*)(wsb + OFF_QT) + (size_t)b * 8192;
  const float* kt = (const float*)(wsb + OFF_KT) + (size_t)b * 8192;
  const float* vt = (const float*)(wsb + OFF_VT) + (size_t)b * 8192;
  int f = tid;
  float e[33];
  for (int j = 0; j < n; ++j) {
    int a = idx[j];
    posembed(skps[m * 3] - skps[a * 3], skps[m * 3 + 1] - skps[a * 3 + 1],
             skps[m * 3 + 2] - skps[a * 3 + 2], e);
    float acc = peb[f];
    for (int d = 0; d < 33; ++d) acc += e[d] * peW[f * 33 + d];
    pe[j * 128 + f] = (h16)acc;
  }
  __syncthreads();
  for (int j = 0; j < n; ++j)
    tb[j * 128 + f] = (h16)(qt[m * 128 + f] - kt[idx[j] * 128 + f] +
                            (float)pe[j * 128 + f]);
  __syncthreads();
  for (int j = 0; j < n; ++j) {
    float acc = b1[f];
    for (int d = 0; d < 128; ++d) acc += (float)tb[j * 128 + d] * W1[f * 128 + d];
    hbuf[j * 128 + f] = (h16)fmaxf(acc, 0.f);
  }
  __syncthreads();
  for (int j = 0; j < n; ++j) {
    float acc = b2[f];
    for (int d = 0; d < 128; ++d) acc += (float)hbuf[j * 128 + d] * W2[f * 128 + d];
    tb[j * 128 + f] = (h16)acc;  // g
  }
  __syncthreads();
  float mx = -3.4e38f;
  for (int j = 0; j < n; ++j) mx = fmaxf(mx, (float)tb[j * 128 + f]);
  float s = 0.f;
  for (int j = 0; j < n; ++j) s += __expf((float)tb[j * 128 + f] - mx);
  float inv = 1.f / s, res = 0.f;
  for (int j = 0; j < n; ++j)
    res += __expf((float)tb[j * 128 + f] - mx) * inv *
           (vt[idx[j] * 128 + f] + (float)pe[j * 128 + f]);
  res += featsIn[(size_t)b * 8192 + m * 128 + f];
  resOut[(size_t)b * 8192 + m * 128 + f] = res;
  float* bn = (float*)(wsb + OFF_BN);
  atomicAdd(&bn[f], res);
  atomicAdd(&bn[128 + f], res * res);
}

__global__ __launch_bounds__(128) void K_bn(const float* resIn,
    const float* bng, const float* bnb, uint8_t* wsb, float* featsOut) {
  int f = threadIdx.x;
  const float* bn = (const float*)(wsb + OFF_BN);
  float mu = bn[f] / 128.f;
  float var = bn[128 + f] / 128.f - mu * mu;
  float sc = rsqrtf(var + 1e-5f) * bng[f];
  float bb = bnb[f];
  for (int r = 0; r < 128; ++r)
    featsOut[r * 128 + f] = (resIn[r * 128 + f] - mu) * sc + bb;
}

// ---------------- K_crossprep: K/V tables + q/kg/vg ----------------
__global__ __launch_bounds__(256) void K_crossprep(
    const float* featsIn, const float* lat, const float* wks, const float* wvs,
    const float* wqs, const float* wkg, const float* wvg,
    const float* gb1, const float* gb2, uint8_t* wsb) {
  int b = blockIdx.x, tid = threadIdx.x;
  __shared__ float sf[8192];
  const float* F = featsIn + (size_t)b * 8192;
  for (int i = tid; i < 8192; i += 256) sf[i] = F[i];
  __syncthreads();
  h16* Kt = (h16*)(wsb + OFF_KTH) + (size_t)b * 8192;
  h16* Vt = (h16*)(wsb + OFF_VTH) + (size_t)b * 8192;
  for (int i = tid; i < 8192; i += 256) {
    int m = i >> 7, f = i & 127;
    float a1 = 0, a2 = 0;
    for (int d = 0; d < 128; ++d) {
      float x = sf[m * 128 + d];
      a1 += x * wks[f * 128 + d];
      a2 += x * wvs[f * 128 + d];
    }
    Kt[i] = (h16)a1; Vt[i] = (h16)a2;
  }
  float* qv = (float*)(wsb + OFF_QV) + b * 128;
  float* kg = (float*)(wsb + OFF_KG) + b * 128;
  float* vg = (float*)(wsb + OFF_VG) + b * 128;
  const float* lb = lat + b * 128;
  for (int f = tid; f < 128; f += 256) {
    float a1 = 0, a2 = 0, a3 = 0;
    for (int d = 0; d < 128; ++d) {
      float x = lb[d];
      a1 += x * wqs[f * 128 + d];
      a2 += x * wkg[f * 128 + d];
      a3 += x * wvg[f * 128 + d];
    }
    qv[f] = a1; kg[f] = a2; vg[f] = a3;
  }
  if (b == 0) {
    float* cb = (float*)(wsb + OFF_CB);
    for (int f = tid; f < 128; f += 256) {
      cb[f] = gb1[f]; cb[128 + f] = gb2[f];
    }
  }
}

// ---------------- K_attn: cross-attention via WMMA ----------------
// 2 waves/block; each wave: 4 queries x 8 slots = 32 rows = 2 WMMA M-tiles.
// Every B fragment is loaded once and reused for both M-tiles.
__global__ __launch_bounds__(64) void K_attn(const float* xyz_q, uint8_t* wsb) {
  __shared__ float s_kps[NA_ * 3];
  __shared__ float s_qv[128], s_kg[128], s_vg[128], s_gb1[128], s_gb2[128];
  __shared__ __align__(16) h16 s_arena[2][14336];  // Xa|pe|t|h per wave
  __shared__ int s_idx[2][4][8];
  __shared__ float s_xy[2][4][2];
  int b = blockIdx.y, tid = threadIdx.x;
  const float* kpsG = (const float*)(wsb + OFF_KPS) + (size_t)b * NA_ * 3;
  for (int i = tid; i < NA_ * 3; i += 64) s_kps[i] = kpsG[i];
  {
    const float* qvG = (const float*)(wsb + OFF_QV) + b * 128;
    const float* kgG = (const float*)(wsb + OFF_KG) + b * 128;
    const float* vgG = (const float*)(wsb + OFF_VG) + b * 128;
    const float* cb = (const float*)(wsb + OFF_CB);
    for (int i = tid; i < 128; i += 64) {
      s_qv[i] = qvG[i]; s_kg[i] = kgG[i]; s_vg[i] = vgG[i];
      s_gb1[i] = cb[i]; s_gb2[i] = cb[128 + i];
    }
  }
  const h16* PEWp = (const h16*)(wsb + OFF_PEW);
  const h16* W1p = (const h16*)(wsb + OFF_W1P);
  const h16* W2p = (const h16*)(wsb + OFF_W2P);
  prefetch_frags(PEWp, 16 * 1024);
  __syncthreads();
  int wave = tid >> 5, lane = tid & 31;
  h16* Xa = s_arena[wave];          // 32 x 64
  h16* pe = Xa + 2048;              // 32 x 128
  h16* tb = Xa + 6144;              // 32 x 128
  h16* hb = Xa + 10240;             // 32 x 128
  int qbase = blockIdx.x * 8 + wave * 4;
  // kNN: lanes 0,8,16,24 each own one of 4 queries
  if ((lane & 7) == 0) {
    int qi = lane >> 3, q = qbase + qi;
    float qx = xyz_q[((size_t)b * Q_ + q) * 2 + 0];
    float qy = xyz_q[((size_t)b * Q_ + q) * 2 + 1];
    s_xy[wave][qi][0] = qx; s_xy[wave][qi][1] = qy;
    float bd[NN_]; int bi[NN_];
    for (int j = 0; j < NN_; ++j) { bd[j] = 3.4e38f; bi[j] = 0; }
    for (int a = 0; a < NA_; ++a) {
      float dx = qx - s_kps[a * 3 + 0], dy = qy - s_kps[a * 3 + 1];
      float d2 = dx * dx + dy * dy;
      if (d2 < bd[NN_ - 1]) {
        int j = NN_ - 1;
        while (j > 0 && bd[j - 1] > d2) { bd[j] = bd[j - 1]; bi[j] = bi[j - 1]; --j; }
        bd[j] = d2; bi[j] = a;
      }
    }
    for (int j = 0; j < NN_; ++j) s_idx[wave][qi][j] = bi[j];
    s_idx[wave][qi][7] = 0;
  }
  __syncthreads();
  // build embed rows: 32 rows x 64. Col 33 = 1.0 rides the pe bias; the
  // global-slot row (n==7) stays all-zero so pe == 0 there exactly.
  {
    int r = lane, qi = r >> 3, n = r & 7;
    float e[34];
    if (n < NN_) {
      int a = s_idx[wave][qi][n];
      posembed(s_xy[wave][qi][0] - s_kps[a * 3 + 0],
               s_xy[wave][qi][1] - s_kps[a * 3 + 1], s_kps[a * 3 + 2], e);
      e[33] = 1.f;
    } else for (int j = 0; j < 34; ++j) e[j] = 0.f;
    for (int j = 0; j < 34; ++j) Xa[r * 64 + j] = (h16)e[j];
    for (int j = 34; j < 64; ++j) Xa[r * 64 + j] = (h16)0.f;
  }
  __syncthreads();
  prefetch_frags(W1p, 32 * 1024);
  // pe = embed @ [peW; pe_b]
  for (int nt = 0; nt < 8; ++nt) {
    v8f a0 = {}, a1 = {};
    for (int kt = 0; kt < 2; ++kt) {
      v16h bb = load_B_frag(PEWp, nt * 2 + kt);
      a0 = wmma(load_A(Xa, 64, kt * 32), bb, a0);
      a1 = wmma(load_A(Xa + 1024, 64, kt * 32), bb, a1);
    }
    store_tile0(pe, 128, nt * 16, a0, false);
    store_tile0(pe + 2048, 128, nt * 16, a1, false);
  }
  __syncthreads();
  // t = q - K[idx] + pe
  const h16* KtG = (const h16*)(wsb + OFF_KTH) + (size_t)b * NA_ * 128;
  for (int i = lane; i < 4096; i += 32) {
    int r = i >> 7, d = i & 127, n = r & 7, qi = r >> 3;
    float kv = (n < NN_) ? (float)KtG[s_idx[wave][qi][n] * 128 + d] : s_kg[d];
    tb[i] = (h16)(s_qv[d] - kv + (float)pe[i]);
  }
  __syncthreads();
  prefetch_frags(W2p, 32 * 1024);
  // h = relu(t @ g_W1 + g_b1)
  for (int nt = 0; nt < 8; ++nt) {
    float bv = s_gb1[nt * 16 + (lane & 15)];
    v8f a0 = {}, a1 = {};
    for (int kt = 0; kt < 4; ++kt) {
      v16h bb = load_B_frag(W1p, nt * 4 + kt);
      a0 = wmma(load_A(tb, 128, kt * 32), bb, a0);
      a1 = wmma(load_A(tb + 2048, 128, kt * 32), bb, a1);
    }
    store_tile(hb, 128, nt * 16, a0, bv, true);
    store_tile(hb + 2048, 128, nt * 16, a1, bv, true);
  }
  __syncthreads();
  // g = h @ g_W2 + g_b2  (stored over t)
  for (int nt = 0; nt < 8; ++nt) {
    float bv = s_gb2[nt * 16 + (lane & 15)];
    v8f a0 = {}, a1 = {};
    for (int kt = 0; kt < 4; ++kt) {
      v16h bb = load_B_frag(W2p, nt * 4 + kt);
      a0 = wmma(load_A(hb, 128, kt * 32), bb, a0);
      a1 = wmma(load_A(hb + 2048, 128, kt * 32), bb, a1);
    }
    store_tile(tb, 128, nt * 16, a0, bv, false);
    store_tile(tb + 2048, 128, nt * 16, a1, bv, false);
  }
  __syncthreads();
  // per-feature softmax over 8 slots; weighted sum of (V + pe) -> loc
  const h16* VtG = (const h16*)(wsb + OFF_VTH) + (size_t)b * NA_ * 128;
  h16* locG = (h16*)(wsb + OFF_LOC);
  for (int i = lane; i < 512; i += 32) {
    int qi = i >> 7, f = i & 127;
    float gv[8], mx = -3.4e38f;
    for (int n = 0; n < 8; ++n) {
      gv[n] = (float)tb[(qi * 8 + n) * 128 + f];
      mx = fmaxf(mx, gv[n]);
    }
    float s = 0.f;
    for (int n = 0; n < 8; ++n) { gv[n] = __expf(gv[n] - mx); s += gv[n]; }
    float inv = 1.f / s, res = 0.f;
    for (int n = 0; n < 8; ++n) {
      float vv = (n < NN_) ? (float)VtG[s_idx[wave][qi][n] * 128 + f] : s_vg[f];
      res += gv[n] * inv * (vv + (float)pe[(qi * 8 + n) * 128 + f]);
    }
    int q = qbase + qi;
    locG[((size_t)b * Q_ + q) * 128 + f] = (h16)res;
  }
}

// ---------------- K_heads: render heads via WMMA ----------------
// grid (Q/128, 3 heads, B); 4 waves/block, wave = 32 queries (2 M-tiles).
__global__ __launch_bounds__(128) void K_heads(const uint8_t* wsb, float* out) {
  __shared__ __align__(16) h16 s_r[4][2048];   // relu(net): 32x64 per wave
  __shared__ __align__(16) h16 s_h2[4][2048];  // relu(h):   32x64 per wave
  int hsel = blockIdx.y, b = blockIdx.z;
  const int chOff = (hsel == 0) ? 0 : (hsel == 1) ? 1 : 2;
  const int odim = (hsel == 2) ? 3 : 1;
  const h16* HW = (const h16*)(wsb + OFF_HEAD + (size_t)hsel * HEAD_STRIDE);
  const float* HB =
      (const float*)(wsb + OFF_HEAD + (size_t)hsel * HEAD_STRIDE + HB_BIAS);
  int wave = threadIdx.x >> 5, lane = threadIdx.x & 31;
  int q0 = blockIdx.x * 128 + wave * 32;
  const h16* y = (const h16*)(wsb + OFF_LOC) + ((size_t)b * Q_ + q0) * 128;
  h16* R = s_r[wave];
  h16* H2 = s_h2[wave];
  int col = lane & 15;
  v8f net[2][4];
  for (int nt = 0; nt < 4; ++nt) {          // net = init(y)
    v8f a0 = {}, a1 = {};
    for (int kt = 0; kt < 4; ++kt) {
      v16h bb = load_B_frag(HW, nt * 4 + kt);
      a0 = wmma(load_A(y, 128, kt * 32), bb, a0);
      a1 = wmma(load_A(y + 2048, 128, kt * 32), bb, a1);
    }
    float bv = HB[nt * 16 + col];
    for (int r = 0; r < 8; ++r) { a0[r] += bv; a1[r] += bv; }
    net[0][nt] = a0; net[1][nt] = a1;
  }
  for (int i = 0; i < 5; ++i) {
    const h16* FC = HW + 8192 + 8192 * i;   // net += fc_i(y)
    prefetch_frags(HW + 49152 + 8192 * i, 16 * 1024);  // W0|W1 of this block
    for (int nt = 0; nt < 4; ++nt) {
      v8f a0 = {}, a1 = {};
      for (int kt = 0; kt < 4; ++kt) {
        v16h bb = load_B_frag(FC, nt * 4 + kt);
        a0 = wmma(load_A(y, 128, kt * 32), bb, a0);
        a1 = wmma(load_A(y + 2048, 128, kt * 32), bb, a1);
      }
      float bv = HB[64 + 64 * i + nt * 16 + col];
      for (int r = 0; r < 8; ++r) { a0[r] += bv; a1[r] += bv; }
      net[0][nt] += a0; net[1][nt] += a1;
    }
    for (int t = 0; t < 2; ++t)
      for (int nt = 0; nt < 4; ++nt)
        store_tile0(R + t * 1024, 64, nt * 16, net[t][nt], true);
    const h16* W0 = HW + 49152 + 8192 * i;  // h = relu(relu(net)@W0+b0)
    for (int nt = 0; nt < 4; ++nt) {
      v8f a0 = {}, a1 = {};
      for (int kt = 0; kt < 2; ++kt) {
        v16h bb = load_B_frag(W0, nt * 2 + kt);
        a0 = wmma(load_A(R, 64, kt * 32), bb, a0);
        a1 = wmma(load_A(R + 1024, 64, kt * 32), bb, a1);
      }
      float bv = HB[384 + 128 * i + nt * 16 + col];
      store_tile(H2, 64, nt * 16, a0, bv, true);
      store_tile(H2 + 1024, 64, nt * 16, a1, bv, true);
    }
    const h16* W1h = HW + 49152 + 8192 * i + 4096;  // net += relu(h)@W1+b1
    for (int nt = 0; nt < 4; ++nt) {
      v8f a0 = net[0][nt], a1 = net[1][nt];
      for (int kt = 0; kt < 2; ++kt) {
        v16h bb = load_B_frag(W1h, nt * 2 + kt);
        a0 = wmma(load_A(H2, 64, kt * 32), bb, a0);
        a1 = wmma(load_A(H2 + 1024, 64, kt * 32), bb, a1);
      }
      float bv = HB[448 + 128 * i + nt * 16 + col];
      for (int r = 0; r < 8; ++r) { a0[r] += bv; a1[r] += bv; }
      net[0][nt] = a0; net[1][nt] = a1;
    }
  }
  for (int t = 0; t < 2; ++t)
    for (int nt = 0; nt < 4; ++nt)
      store_tile0(R + t * 1024, 64, nt * 16, net[t][nt], true);
  {
    v8f a0 = {}, a1 = {};
    const h16* WO = HW + 90112;
    for (int kt = 0; kt < 2; ++kt) {
      v16h bb = load_B_frag(WO, kt);
      a0 = wmma(load_A(R, 64, kt * 32), bb, a0);
      a1 = wmma(load_A(R + 1024, 64, kt * 32), bb, a1);
    }
    float bv = HB[1024 + col];
    if (col < odim) {
      int mb = (lane >> 4) * 8;
      for (int r = 0; r < 8; ++r) {
        out[((size_t)b * Q_ + q0 + mb + r) * 5 + chOff + col] = a0[r] + bv;
        out[((size_t)b * Q_ + q0 + 16 + mb + r) * 5 + chOff + col] = a1[r] + bv;
      }
    }
  }
}

// ---------------- host ----------------
extern "C" void kernel_launch(void* const* d_in, const int* in_sizes, int n_in,
                              void* d_out, int out_size, void* d_ws,
                              size_t ws_size, hipStream_t stream) {
  (void)in_sizes; (void)out_size; (void)ws_size;
  uint8_t* ws = (uint8_t*)d_ws;
  Ptrs P{};
  int nc = n_in < 160 ? n_in : 160;
  for (int i = 0; i < nc; ++i) P.p[i] = (const float*)d_in[i];
  float* out = (float*)d_out;
  float* FA = (float*)(ws + OFF_FA);
  float* FB = (float*)(ws + OFF_FB);

  hipLaunchKernelGGL(K_prep0, dim3(1), dim3(128), 0, stream,
                     P.p[1], P.p[85], P.p[86], P.p[2], P.p[4], P.p[5], ws);
  hipLaunchKernelGGL(K_pack, dim3(64), dim3(256), 0, stream, P, ws);

  auto runT = [&](int tb, const float* fin, int knnK, float* fout) {
    hipLaunchKernelGGL(K_tab, dim3(2), dim3(256), 0, stream, fin,
                       P.p[tb + 9], P.p[tb + 8], P.p[tb + 10], ws);
    hipLaunchKernelGGL(K_tattn, dim3(128), dim3(128), 0, stream, fin,
                       P.p[tb + 6], P.p[tb + 7], P.p[tb + 2], P.p[tb + 4],
                       P.p[tb + 3], P.p[tb + 5], ws, FB, knnK);
    hipLaunchKernelGGL(K_bn, dim3(1), dim3(128), 0, stream, FB,
                       P.p[tb + 1], P.p[tb + 0], ws, fout);
  };
  runT(121, P.p[3], 8, FA);   // t1 (kNN=8) on anchor_feats
  runT(132, FA, 64, FA);      // t2 (group_all)
  runT(143, FA, 64, FA);      // t3 (group_all)

  const int cb = 40;          // ct1
  hipLaunchKernelGGL(K_crossprep, dim3(2), dim3(256), 0, stream, FA,
                     (const float*)(ws + OFF_LAT), P.p[cb + 7], P.p[cb + 10],
                     P.p[cb + 8], P.p[cb + 6], P.p[cb + 9],
                     P.p[cb + 2], P.p[cb + 3], ws);

  hipLaunchKernelGGL(K_attn, dim3(Q_ / 8, B_), dim3(64), 0, stream,
                     P.p[0], ws);
  hipLaunchKernelGGL(K_heads, dim3(Q_ / 128, 3, B_), dim3(128), 0, stream,
                     ws, out);
}
